// EGNN_combined_graph_67800353734915
// MI455X (gfx1250) — matrix operand order repaired and checked by reading, not compiled, not run on hardware
//
#include <hip/hip_runtime.h>
#include <hip/hip_bf16.h>

typedef __attribute__((ext_vector_type(16))) _Float16 v16h;
typedef __attribute__((ext_vector_type(8)))  _Float16 v8h;
typedef __attribute__((ext_vector_type(8)))  float    v8f;

constexpr int NTOT   = 4096;
constexpr int NPER   = 256;
constexpr int NGRAPH = 16;
constexpr int DIN    = 16;
constexpr int HD     = 128;
constexpr int LAY    = 4;
constexpr int KNB    = 32;   // knn k
constexpr int NGAUSS = 20;
constexpr int FE     = 280;  // 2H + G + 4

// ---------------------------------------------------------------- helpers
__device__ __forceinline__ float silu_f(float v)    { return v * (1.0f / (1.0f + __expf(-v))); }
__device__ __forceinline__ float sigmoid_f(float v) { return 1.0f / (1.0f + __expf(-v)); }

// A-fragment (16x32 f16, ISA 7.12.2 layout) from row-major f16 source.
// lane<16: row=lane&15, K = kc + {0..7, 16..23}; lane>=16: K offset +8.
__device__ __forceinline__ v16h load_afrag(const _Float16* base, int row, int stride,
                                           int kc, int lane) {
  int khalf = (lane >> 4) & 1;
  const _Float16* p = base + row * stride + kc + khalf * 8;
  v8h lo = *(const v8h*)p;
  v8h hi = *(const v8h*)(p + 16);
  v16h a;
#pragma unroll
  for (int i = 0; i < 8; ++i) { a[i] = lo[i]; a[i + 8] = hi[i]; }
  return a;
}

// B-fragment: pre-swizzled weights, one contiguous 32B load per lane.
__device__ __forceinline__ v16h load_bfrag(const _Float16* Wp, int kci, int nb, int lane) {
  return *(const v16h*)(Wp + (((kci << 3) + nb) << 9) + (lane << 4));
}

// ---------------------------------------------------------------- weight pre-swizzle
// dst fragment layout: frag=(kb,nb), lane, j :  K = kb*32 + ((j&8)<<1) + ((lane&16)>>1) + (j&7) ; col = nb*16 + (lane&15)
// K >= KDsrc rows are zero-padded (used for the 24-row gauss+onehot tail tile).
__global__ __launch_bounds__(256) void pack_kernel(const float* __restrict__ src,
                                                   _Float16* __restrict__ dst,
                                                   int KD, int layerStrideSrc, int KDsrc) {
  int o   = blockIdx.x * 256 + threadIdx.x;
  int per = KD * HD;
  int l   = o / per;
  int oo  = o - l * per;
  int frag = oo >> 9, idx = oo & 511, lane = idx >> 4, j = idx & 15;
  int kb = frag >> 3, nb = frag & 7;
  int K   = kb * 32 + ((j & 8) << 1) + ((lane & 16) >> 1) + (j & 7);
  int col = nb * 16 + (lane & 15);
  float v = (K < KDsrc) ? src[(size_t)l * layerStrideSrc + K * HD + col] : 0.0f;
  dst[o] = (_Float16)v;
}

// ---------------------------------------------------------------- input embedding
__global__ __launch_bounds__(128) void embed_kernel(const float* __restrict__ h_in,
                                                    const float* __restrict__ x_in,
                                                    const float* __restrict__ t_in,
                                                    const float* __restrict__ W_in,
                                                    const float* __restrict__ b_in,
                                                    float* __restrict__ h_cur,
                                                    _Float16* __restrict__ h16,
                                                    float* __restrict__ x_cur) {
  int i = blockIdx.x, c = threadIdx.x;
  float acc = b_in[c];
#pragma unroll
  for (int d = 0; d < DIN; ++d) acc += h_in[i * DIN + d] * W_in[d * HD + c];
  acc += t_in[i] * W_in[DIN * HD + c];
  h_cur[i * HD + c] = acc;
  h16[i * HD + c]   = (_Float16)acc;
  if (c < 3) x_cur[i * 3 + c] = x_in[i * 3 + c];
}

// ---------------------------------------------------------------- knn (per graph)
__global__ __launch_bounds__(256) void knn_kernel(const float* __restrict__ xcur,
                                                  int* __restrict__ nbr) {
  __shared__ float xs[NPER * 3];
  int g = blockIdx.x, t = threadIdx.x;
  int base = g * NPER;
  for (int i = t; i < NPER * 3; i += NPER) xs[i] = xcur[base * 3 + i];
  __syncthreads();
  float xi0 = xs[t * 3 + 0], xi1 = xs[t * 3 + 1], xi2 = xs[t * 3 + 2];
  unsigned sel[8] = {0, 0, 0, 0, 0, 0, 0, 0};
  sel[t >> 5] |= 1u << (t & 31);                 // exclude self
  for (int k = 0; k < KNB; ++k) {
    float best = 3.0e38f; int bj = 0;
    for (int j = 0; j < NPER; ++j) {
      if ((sel[j >> 5] >> (j & 31)) & 1u) continue;
      float d0 = xi0 - xs[j * 3 + 0];
      float d1 = xi1 - xs[j * 3 + 1];
      float d2 = xi2 - xs[j * 3 + 2];
      float d  = d0 * d0 + d1 * d1 + d2 * d2;
      if (d < best) { best = d; bj = j; }
    }
    sel[bj >> 5] |= 1u << (bj & 31);
    nbr[(base + t) * KNB + k] = base + bj;
  }
}

// ---------------------------------------------------------------- per-node GEMM: out = h16 @ Wp (+bias)
__global__ __launch_bounds__(64) void gemm_node_kernel(const _Float16* __restrict__ h16,
                                                       const _Float16* __restrict__ Wp,
                                                       const float* __restrict__ bias,
                                                       float* __restrict__ out) {
  int lane = threadIdx.x & 31, w = threadIdx.x >> 5;
  int m0 = blockIdx.x * 32 + w * 16;
  v8f zero = {};
  v8f acc[8];
#pragma unroll
  for (int nb = 0; nb < 8; ++nb) acc[nb] = zero;
#pragma unroll
  for (int kci = 0; kci < 4; ++kci) {
    v16h a = load_afrag(h16, m0 + (lane & 15), HD, kci * 32, lane);
#pragma unroll
    for (int nb = 0; nb < 8; ++nb) {
      v16h b = load_bfrag(Wp, kci, nb, lane);
      acc[nb] = __builtin_amdgcn_wmma_f32_16x16x32_f16(false, a, false, b, (short)0,
                                                       acc[nb], false, false);
    }
  }
#pragma unroll
  for (int nb = 0; nb < 8; ++nb) {
    int c = nb * 16 + (lane & 15);
    float bv = bias ? bias[c] : 0.0f;
#pragma unroll
    for (int r = 0; r < 8; ++r) {
      int row = m0 + r + 8 * (lane >> 4);
      out[row * HD + c] = acc[nb][r] + bv;
    }
  }
}

// ---------------------------------------------------------------- fused edge pipeline (1 block = 1 node, 32 edges)
__global__ __launch_bounds__(64) void edge_kernel(const float* __restrict__ x_cur,
                                                  const float* __restrict__ A,
                                                  const float* __restrict__ B,
                                                  const int* __restrict__ nbr,
                                                  const int* __restrict__ gmask,
                                                  const _Float16* __restrict__ Wtp,   // packed gauss+onehot tail (32x128)
                                                  const _Float16* __restrict__ We2p,
                                                  const float* __restrict__ be2,
                                                  const float* __restrict__ Wei,
                                                  const float* __restrict__ bei,
                                                  const _Float16* __restrict__ Wx1p,
                                                  const float* __restrict__ bx1,
                                                  const float* __restrict__ Wx2,
                                                  float* __restrict__ mi_out,
                                                  float* __restrict__ dx_out) {
  __shared__ _Float16 tail_h[32 * 32];  // [dfeat(20), onehot(4), 0 x 8] per edge
  __shared__ int      srcs_s[32];
  __shared__ _Float16 u_h[32 * HD];     // silu(feat@We1+be1) as f16
  __shared__ float    mij_f[32 * HD];
  __shared__ _Float16 mij_h[32 * HD];
  __shared__ float    eij_s[32];
  __shared__ float    relw_s[32 * 3];   // rel/(dist+1)
  __shared__ float    xw_s[32];

  int node = blockIdx.x;
  int t = threadIdx.x, lane = t & 31, w = t >> 5;

  // ---- phase A: per-edge scalars (distance, gaussian features, edge-type one-hot)
  if (t < 32) {
    int k = t;
    int src = nbr[node * KNB + k];
    srcs_s[k] = src;
    float rx = x_cur[node * 3 + 0] - x_cur[src * 3 + 0];
    float ry = x_cur[node * 3 + 1] - x_cur[src * 3 + 1];
    float rz = x_cur[node * 3 + 2] - x_cur[src * 3 + 2];
    float d2 = rx * rx + ry * ry + rz * rz;
    float dist = sqrtf(d2 + 1e-8f);
    const float step = 10.0f / 19.0f;
    const float coeff = -0.5f / (step * step);
#pragma unroll
    for (int g = 0; g < NGAUSS; ++g) {
      float dd = dist - step * (float)g;
      tail_h[k * 32 + g] = (_Float16)__expf(coeff * dd * dd);
    }
    int ms = (gmask[src] != 0), md = (gmask[node] != 0);
    int etype = (1 - ms) * 2 + (1 - md);
#pragma unroll
    for (int e = 0; e < 4; ++e)
      tail_h[k * 32 + NGAUSS + e] = (e == etype) ? (_Float16)1.0f : (_Float16)0.0f;
#pragma unroll
    for (int z = 0; z < 8; ++z) tail_h[k * 32 + 24 + z] = (_Float16)0.0f;
    float inv = 1.0f / (dist + 1.0f);
    relw_s[k * 3 + 0] = rx * inv;
    relw_s[k * 3 + 1] = ry * inv;
    relw_s[k * 3 + 2] = rz * inv;
  }
  __syncthreads();

  // ---- GEMM0 (tail WMMA, K=32) + gather writeback: u = silu(A[dst] + B[src] + tail@Wtail)
  {
    int m0 = w * 16;
    v8f zero = {};
    v8f acc[8];
    v16h a0 = load_afrag(tail_h, m0 + (lane & 15), 32, 0, lane);
#pragma unroll
    for (int nb = 0; nb < 8; ++nb) {
      v16h b = load_bfrag(Wtp, 0, nb, lane);
      acc[nb] = __builtin_amdgcn_wmma_f32_16x16x32_f16(false, a0, false, b, (short)0,
                                                       zero, false, false);
    }
#pragma unroll
    for (int nb = 0; nb < 8; ++nb) {
      int c = nb * 16 + (lane & 15);
      float av = A[node * HD + c];
#pragma unroll
      for (int r = 0; r < 8; ++r) {
        int m = m0 + r + 8 * (lane >> 4);
        int src = srcs_s[m];
        float v = acc[nb][r] + av + B[src * HD + c];
        u_h[m * HD + c] = (_Float16)silu_f(v);
      }
    }
  }
  __syncthreads();

  // ---- GEMM1: mij = silu(u @ We2 + be2)   (32x128 @ 128x128, f16 WMMA)
  {
    int m0 = w * 16;
    v8f zero = {};
    v8f acc[8];
#pragma unroll
    for (int nb = 0; nb < 8; ++nb) acc[nb] = zero;
#pragma unroll
    for (int kci = 0; kci < 4; ++kci) {
      v16h a = load_afrag(u_h, m0 + (lane & 15), HD, kci * 32, lane);
#pragma unroll
      for (int nb = 0; nb < 8; ++nb) {
        v16h b = load_bfrag(We2p, kci, nb, lane);
        acc[nb] = __builtin_amdgcn_wmma_f32_16x16x32_f16(false, a, false, b, (short)0,
                                                         acc[nb], false, false);
      }
    }
#pragma unroll
    for (int nb = 0; nb < 8; ++nb) {
      int c = nb * 16 + (lane & 15);
      float bv = be2[c];
#pragma unroll
      for (int r = 0; r < 8; ++r) {
        int m = m0 + r + 8 * (lane >> 4);
        float v = silu_f(acc[nb][r] + bv);
        mij_f[m * HD + c] = v;
        mij_h[m * HD + c] = (_Float16)v;
      }
    }
  }
  __syncthreads();

  // ---- eij = sigmoid(mij . Wei + bei)
  if (t < 32) {
    float s = bei[0];
    for (int c = 0; c < HD; ++c) s += mij_f[t * HD + c] * Wei[c];
    eij_s[t] = sigmoid_f(s);
  }
  __syncthreads();

  // ---- mi[node] = sum_k mij[k]*eij[k]   (contiguous dst segment, no atomics)
  {
    int c0 = t * 2;
    float s0 = 0.0f, s1 = 0.0f;
    for (int k = 0; k < KNB; ++k) {
      float e = eij_s[k];
      s0 += mij_f[k * HD + c0 + 0] * e;
      s1 += mij_f[k * HD + c0 + 1] * e;
    }
    mi_out[node * HD + c0 + 0] = s0;
    mi_out[node * HD + c0 + 1] = s1;
  }

  // ---- GEMM2: xw = tanh( silu(mij @ Wx1 + bx1) . Wx2 )
  {
    int m0 = w * 16;
    v8f zero = {};
    v8f acc[8];
#pragma unroll
    for (int nb = 0; nb < 8; ++nb) acc[nb] = zero;
#pragma unroll
    for (int kci = 0; kci < 4; ++kci) {
      v16h a = load_afrag(mij_h, m0 + (lane & 15), HD, kci * 32, lane);
#pragma unroll
      for (int nb = 0; nb < 8; ++nb) {
        v16h b = load_bfrag(Wx1p, kci, nb, lane);
        acc[nb] = __builtin_amdgcn_wmma_f32_16x16x32_f16(false, a, false, b, (short)0,
                                                         acc[nb], false, false);
      }
    }
    float part[8];
#pragma unroll
    for (int r = 0; r < 8; ++r) part[r] = 0.0f;
#pragma unroll
    for (int nb = 0; nb < 8; ++nb) {
      int c = nb * 16 + (lane & 15);
      float wx2 = Wx2[c];
      float bb  = bx1[c];
#pragma unroll
      for (int r = 0; r < 8; ++r) part[r] += silu_f(acc[nb][r] + bb) * wx2;
    }
#pragma unroll
    for (int r = 0; r < 8; ++r) {
      float p = part[r];                         // reduce 16 lanes sharing row m
      p += __shfl_xor(p, 1, 32);
      p += __shfl_xor(p, 2, 32);
      p += __shfl_xor(p, 4, 32);
      p += __shfl_xor(p, 8, 32);
      if ((lane & 15) == 0) {
        int m = m0 + r + 8 * (lane >> 4);
        xw_s[m] = tanhf(p);
      }
    }
  }
  __syncthreads();

  // ---- dx[node] = sum_k relw[k] * xw[k]
  if (t < 3) {
    float s = 0.0f;
    for (int k = 0; k < KNB; ++k) s += relw_s[k * 3 + t] * xw_s[k];
    dx_out[node * 3 + t] = s;
  }
}

// ---------------------------------------------------------------- node MLP + residual + coord update
__global__ __launch_bounds__(64) void node_kernel(const float* __restrict__ mi,
                                                  const float* __restrict__ dxv,
                                                  const int* __restrict__ gmask,
                                                  const _Float16* __restrict__ Wn1p,
                                                  const float* __restrict__ bn1,
                                                  const _Float16* __restrict__ Wn2p,
                                                  const float* __restrict__ bn2,
                                                  float* __restrict__ h_cur,
                                                  _Float16* __restrict__ h16,
                                                  float* __restrict__ x_cur) {
  __shared__ _Float16 cat_h[32 * 256];
  __shared__ _Float16 t1_h[32 * HD];
  int t = threadIdx.x, lane = t & 31, w = t >> 5;
  int r0 = blockIdx.x * 32;

  if (t < 32) {                                   // x += dx * mask
    int row = r0 + t;
    float mf = gmask[row] ? 1.0f : 0.0f;
#pragma unroll
    for (int c = 0; c < 3; ++c) x_cur[row * 3 + c] += dxv[row * 3 + c] * mf;
  }
  for (int e = t; e < 32 * 256; e += 64) {        // stage [mi, h] as f16
    int row = e >> 8, c = e & 255;
    float v = (c < HD) ? mi[(r0 + row) * HD + c] : h_cur[(r0 + row) * HD + (c - HD)];
    cat_h[e] = (_Float16)v;
  }
  __syncthreads();

  int m0 = w * 16;
  v8f zero = {};
  {
    v8f acc[8];
#pragma unroll
    for (int nb = 0; nb < 8; ++nb) acc[nb] = zero;
#pragma unroll
    for (int kci = 0; kci < 8; ++kci) {           // KD = 256
      v16h a = load_afrag(cat_h, m0 + (lane & 15), 256, kci * 32, lane);
#pragma unroll
      for (int nb = 0; nb < 8; ++nb) {
        v16h b = load_bfrag(Wn1p, kci, nb, lane);
        acc[nb] = __builtin_amdgcn_wmma_f32_16x16x32_f16(false, a, false, b, (short)0,
                                                         acc[nb], false, false);
      }
    }
#pragma unroll
    for (int nb = 0; nb < 8; ++nb) {
      int c = nb * 16 + (lane & 15);
      float bv = bn1[c];
#pragma unroll
      for (int r = 0; r < 8; ++r) {
        int m = m0 + r + 8 * (lane >> 4);
        t1_h[m * HD + c] = (_Float16)silu_f(acc[nb][r] + bv);
      }
    }
  }
  __syncthreads();
  {
    v8f acc[8];
#pragma unroll
    for (int nb = 0; nb < 8; ++nb) acc[nb] = zero;
#pragma unroll
    for (int kci = 0; kci < 4; ++kci) {           // KD = 128
      v16h a = load_afrag(t1_h, m0 + (lane & 15), HD, kci * 32, lane);
#pragma unroll
      for (int nb = 0; nb < 8; ++nb) {
        v16h b = load_bfrag(Wn2p, kci, nb, lane);
        acc[nb] = __builtin_amdgcn_wmma_f32_16x16x32_f16(false, a, false, b, (short)0,
                                                         acc[nb], false, false);
      }
    }
#pragma unroll
    for (int nb = 0; nb < 8; ++nb) {
      int c = nb * 16 + (lane & 15);
      float bv = bn2[c];
#pragma unroll
      for (int r = 0; r < 8; ++r) {
        int m = m0 + r + 8 * (lane >> 4);
        int row = r0 + m;
        float mf = gmask[row] ? 1.0f : 0.0f;
        float hv = h_cur[row * HD + c] + (acc[nb][r] + bv) * mf;
        h_cur[row * HD + c] = hv;
        h16[row * HD + c]   = (_Float16)hv;
      }
    }
  }
}

// ---------------------------------------------------------------- output head
__global__ __launch_bounds__(32) void out_kernel(const float* __restrict__ h_cur,
                                                 const float* __restrict__ x_cur,
                                                 const float* __restrict__ h0,
                                                 const int* __restrict__ gmask,
                                                 const float* __restrict__ W_out,
                                                 const float* __restrict__ b_out,
                                                 float* __restrict__ out) {
  int i = blockIdx.x, t = threadIdx.x;
  float mf = gmask[i] ? 1.0f : 0.0f;
  if (t < DIN) {
    float s = b_out[t];
    for (int k = 0; k < HD; ++k) s += h_cur[i * HD + k] * W_out[k * DIN + t];
    out[i * DIN + t] = s * mf + h0[i * DIN + t] * (1.0f - mf);
  } else if (t < DIN + 3) {
    int c = t - DIN;
    out[NTOT * DIN + i * 3 + c] = x_cur[i * 3 + c];
  }
}

// ---------------------------------------------------------------- launcher
extern "C" void kernel_launch(void* const* d_in, const int* in_sizes, int n_in,
                              void* d_out, int out_size, void* d_ws, size_t ws_size,
                              hipStream_t stream) {
  (void)in_sizes; (void)n_in; (void)out_size; (void)ws_size;
  const float* h_in  = (const float*)d_in[0];
  const float* x_in  = (const float*)d_in[1];
  const float* t_in  = (const float*)d_in[2];
  const int*   gmask = (const int*)d_in[3];
  const float* W_in  = (const float*)d_in[5];
  const float* b_in  = (const float*)d_in[6];
  const float* W_out = (const float*)d_in[7];
  const float* b_out = (const float*)d_in[8];
  const float* We1   = (const float*)d_in[9];
  const float* be1   = (const float*)d_in[10];
  const float* We2   = (const float*)d_in[11];
  const float* be2   = (const float*)d_in[12];
  const float* Wei   = (const float*)d_in[13];
  const float* bei   = (const float*)d_in[14];
  const float* Wn1   = (const float*)d_in[15];
  const float* bn1   = (const float*)d_in[16];
  const float* Wn2   = (const float*)d_in[17];
  const float* bn2   = (const float*)d_in[18];
  const float* Wx1   = (const float*)d_in[19];
  const float* bx1   = (const float*)d_in[20];
  const float* Wx2   = (const float*)d_in[21];

  char* ws = (char*)d_ws;
  float*    h_cur = (float*)   (ws + 0);
  float*    Abuf  = (float*)   (ws + 2097152);
  float*    Bbuf  = (float*)   (ws + 4194304);
  float*    mi    = (float*)   (ws + 6291456);
  _Float16* h16   = (_Float16*)(ws + 8388608);
  _Float16* We1dp = (_Float16*)(ws + 9437184);
  _Float16* We1sp = (_Float16*)(ws + 9568256);
  _Float16* We2p  = (_Float16*)(ws + 9699328);
  _Float16* Wx1p  = (_Float16*)(ws + 9830400);
  _Float16* Wn2p  = (_Float16*)(ws + 9961472);
  _Float16* Wn1p  = (_Float16*)(ws + 10092544);
  float*    x_cur = (float*)   (ws + 10354688);
  float*    dxb   = (float*)   (ws + 10403840);
  int*      nbr   = (int*)     (ws + 10452992);
  _Float16* Wtp   = (_Float16*)(ws + 10977280);   // packed gauss+onehot tail, 32x128/layer

  // pre-swizzle all WMMA weights into B-fragment layout (f16)
  pack_kernel<<<LAY * HD * HD / 256, 256, 0, stream>>>(We1,            We1dp, HD,  FE * HD, HD);
  pack_kernel<<<LAY * HD * HD / 256, 256, 0, stream>>>(We1 + HD * HD,  We1sp, HD,  FE * HD, HD);
  pack_kernel<<<LAY * HD * HD / 256, 256, 0, stream>>>(We2,            We2p,  HD,  HD * HD, HD);
  pack_kernel<<<LAY * HD * HD / 256, 256, 0, stream>>>(Wx1,            Wx1p,  HD,  HD * HD, HD);
  pack_kernel<<<LAY * HD * HD / 256, 256, 0, stream>>>(Wn2,            Wn2p,  HD,  HD * HD, HD);
  pack_kernel<<<LAY * 2 * HD * HD / 256, 256, 0, stream>>>(Wn1,        Wn1p,  256, 2 * HD * HD, 256);
  pack_kernel<<<LAY * 32 * HD / 256, 256, 0, stream>>>(We1 + 2 * HD * HD, Wtp, 32, FE * HD,
                                                       NGAUSS + 4);     // zero-pad rows 24..31

  embed_kernel<<<NTOT, HD, 0, stream>>>(h_in, x_in, t_in, W_in, b_in, h_cur, h16, x_cur);

  for (int l = 0; l < LAY; ++l) {
    knn_kernel<<<NGRAPH, NPER, 0, stream>>>(x_cur, nbr);
    gemm_node_kernel<<<NTOT / 32, 64, 0, stream>>>(h16, We1dp + l * HD * HD, be1 + l * HD, Abuf);
    gemm_node_kernel<<<NTOT / 32, 64, 0, stream>>>(h16, We1sp + l * HD * HD, nullptr,      Bbuf);
    edge_kernel<<<NTOT, 64, 0, stream>>>(
        x_cur, Abuf, Bbuf, nbr, gmask,
        Wtp + l * 32 * HD,
        We2p + l * HD * HD, be2 + l * HD,
        Wei + l * HD, bei + l,
        Wx1p + l * HD * HD, bx1 + l * HD, Wx2 + l * HD,
        mi, dxb);
    node_kernel<<<NTOT / 32, 64, 0, stream>>>(
        mi, dxb, gmask,
        Wn1p + (size_t)l * 2 * HD * HD, bn1 + l * HD,
        Wn2p + l * HD * HD, bn2 + l * HD,
        h_cur, h16, x_cur);
  }

  out_kernel<<<NTOT, 32, 0, stream>>>(h_cur, x_cur, h_in, gmask, W_out, b_out, (float*)d_out);
}
